// gat_encode_51539607552901
// MI455X (gfx1250) — matrix operand order repaired and checked by reading, not compiled, not run on hardware
//
#include <hip/hip_runtime.h>
#include <hip/hip_bf16.h>
#include <math.h>

typedef __attribute__((ext_vector_type(16))) __bf16 v16bf;
typedef __attribute__((ext_vector_type(8)))  float  v8f;
typedef __attribute__((ext_vector_type(4)))  int    v4i;

union BF16Frag { v16bf v; unsigned u[8]; unsigned short h[16]; };
union F32Acc   { v8f v; float f[8]; };

#define Bdim 8
#define Ndim 1024
#define Hdim 512
#define DK   128

// ---- CDNA5 async global->LDS path (guarded; falls back to uint4 copy) -----
#if defined(__has_builtin)
#if __has_builtin(__builtin_amdgcn_global_load_async_to_lds_b128) && \
    __has_builtin(__builtin_amdgcn_s_wait_asynccnt)
#define USE_ASYNC_LDS 1
#endif
#endif

#ifdef USE_ASYNC_LDS
typedef v4i __attribute__((address_space(1)))* gv4i_p;  // global int4*
typedef v4i __attribute__((address_space(3)))* lv4i_p;  // LDS int4*
#endif

__device__ __forceinline__ void cp16_to_lds(unsigned short* dst,
                                            const unsigned short* src) {
#ifdef USE_ASYNC_LDS
  __builtin_amdgcn_global_load_async_to_lds_b128((gv4i_p)src, (lv4i_p)dst, 0, 0);
#else
  *(uint4*)dst = *(const uint4*)src;
#endif
}

// fp32 -> bf16 round-to-nearest-even
__device__ __forceinline__ unsigned short f2bf(float f) {
  unsigned u = __float_as_uint(f);
  unsigned r = ((u >> 16) & 1u) + 0x7FFFu;
  return (unsigned short)((u + r) >> 16);
}
// pack two fp32 -> bf16x2 in one dword
__device__ __forceinline__ unsigned f2bf2(float lo, float hi) {
  return (unsigned)f2bf(lo) | ((unsigned)f2bf(hi) << 16);
}

// A-matrix 16x32 bf16 per-lane K index of pair p (contiguous, matches B):
// lanes 0-15 (hi=0): pairs at K=0,2,...,14
// lanes 16-31 (hi=1): pairs at K=16,18,...,30
__device__ __forceinline__ int a_pair_k(int p, int hi) {
  return hi * 16 + 2 * p;
}

__device__ __forceinline__ unsigned ld_u32(const unsigned short* p) {
  return *(const unsigned*)p;
}

// ---------------------------------------------------------------------------
// Kernel 1: QKV projection.  out = to_bf16(nodes @ W).
// WHICH=0/1: row-major [bh][n][dk] (Q,K);  WHICH=2: transposed [bh][dk][n] (V).
// grid: (rows/64=128, cols/64=8), block 128 (4 waves), each wave: 16x64.
// Software-pipelined: global loads for tile k+32 issued under tile k's WMMAs.
// ---------------------------------------------------------------------------
template <int WHICH>
__global__ __launch_bounds__(128) void qkv_proj_kernel(
    const float* __restrict__ nodes, const float* __restrict__ W,
    unsigned short* __restrict__ outp)
{
  __shared__ unsigned short lx[64 * 32];   // X tile [row64][k32] bf16
  __shared__ unsigned short lw[64 * 32];   // W tile transposed [n64][k32] bf16

  const int t    = threadIdx.x;
  const int lane = t & 31;
  const int w    = t >> 5;
  const int mloc = lane & 15;
  const int hi   = lane >> 4;

  const int rowbase = blockIdx.x * 64;
  const int c0      = blockIdx.y * 64;

  // per-thread staging slices (constant across k)
  const int xr = t >> 3, xc = (t & 7) * 4;    // X: rows 0..15 (+16i), 8 f4/row
  const int wr = t >> 4, wc = (t & 15) * 4;   // W: rows 0..7 (+8i), 16 f4/row

  F32Acc acc[4];
  for (int i = 0; i < 4; ++i)
    for (int r = 0; r < 8; ++r) acc[i].f[r] = 0.0f;

  unsigned* lxu = (unsigned*)lx;

  float4 xreg[4], wreg[4];
  for (int i = 0; i < 4; ++i) {
    xreg[i] = *(const float4*)(nodes + (size_t)(rowbase + xr + i * 16) * Hdim + xc);
    wreg[i] = *(const float4*)(W + (size_t)(wr + i * 8) * Hdim + c0 + wc);
  }

  for (int k = 0; k < Hdim; k += 32) {
    // convert+store staged registers into LDS
    for (int i = 0; i < 4; ++i) {
      const int r = xr + i * 16;
      lxu[(r * 32 + xc) >> 1]       = f2bf2(xreg[i].x, xreg[i].y);
      lxu[((r * 32 + xc) >> 1) + 1] = f2bf2(xreg[i].z, xreg[i].w);
      const int rw = wr + i * 8;
      lw[(wc + 0) * 32 + rw] = f2bf(wreg[i].x);
      lw[(wc + 1) * 32 + rw] = f2bf(wreg[i].y);
      lw[(wc + 2) * 32 + rw] = f2bf(wreg[i].z);
      lw[(wc + 3) * 32 + rw] = f2bf(wreg[i].w);
    }
    __syncthreads();

    // issue next tile's global loads (overlap with WMMA below)
    if (k + 32 < Hdim) {
      for (int i = 0; i < 4; ++i) {
        xreg[i] = *(const float4*)(nodes + (size_t)(rowbase + xr + i * 16) * Hdim + (k + 32) + xc);
        wreg[i] = *(const float4*)(W + (size_t)(k + 32 + wr + i * 8) * Hdim + c0 + wc);
      }
    }

    BF16Frag a;
    for (int p = 0; p < 8; ++p)
      a.u[p] = ld_u32(&lx[(w * 16 + mloc) * 32 + a_pair_k(p, hi)]);
    for (int ti = 0; ti < 4; ++ti) {
      BF16Frag bf;
      const int n = ti * 16 + mloc;
      for (int p = 0; p < 8; ++p)
        bf.u[p] = ld_u32(&lw[n * 32 + hi * 16 + 2 * p]);
      acc[ti].v = __builtin_amdgcn_wmma_f32_16x16x32_bf16(
          false, a.v, false, bf.v, (short)0, acc[ti].v, false, false);
    }
    __syncthreads();
  }

  // store bf16, remapped to per-head layout (straight-line: WHICH is static)
  for (int ti = 0; ti < 4; ++ti) {
    const int col  = c0 + ti * 16 + mloc;
    const int head = col >> 7, dd = col & 127;
    for (int r = 0; r < 8; ++r) {
      const int row = rowbase + w * 16 + (hi ? 8 + r : r);
      const int b = row >> 10, n = row & 1023;
      const unsigned short val = f2bf(acc[ti].f[r]);
      const size_t bh = (size_t)(b * 4 + head);
      if (WHICH == 2) outp[(bh * DK + dd) * Ndim + n] = val;   // V transposed
      else            outp[(bh * Ndim + n) * DK + dd] = val;   // Q/K row-major
    }
  }
}

// issue one tile's worth of async K/V^T staging (8 b128 per thread)
__device__ __forceinline__ void stage_tiles(int kb, unsigned short* lkbuf,
                                            unsigned short* lvbuf,
                                            const unsigned short* Kb,
                                            const unsigned short* Vtb, int t) {
  for (int i = 0; i < 4; ++i) {
    const int e = t + i * 128;
    cp16_to_lds(lkbuf + e * 8, Kb + (size_t)kb * DK + e * 8);
  }
  for (int i = 0; i < 4; ++i) {
    const int e = t + i * 128;
    const int d = e >> 2, q = e & 3;
    cp16_to_lds(lvbuf + e * 8, Vtb + (size_t)d * Ndim + kb + q * 8);
  }
}

// ---------------------------------------------------------------------------
// Kernel 2: masked flash attention + ReLU epilogue.
// grid: (N/64=16, B*HEADS=32), block 128 (4 waves); wave w: 16 query rows.
// Double-buffered async global->LDS staging; row-sum of P via ones-WMMA.
// ---------------------------------------------------------------------------
__global__ __launch_bounds__(128) void gat_attn_kernel(
    const unsigned short* __restrict__ Qm,
    const unsigned short* __restrict__ Km,
    const unsigned short* __restrict__ Vtm,
    const int* __restrict__ adjs,
    float* __restrict__ out)
{
  __shared__ unsigned short lk[2][32 * 128];    // K tiles [key32][ch128]
  __shared__ unsigned short lvt[2][128 * 32];   // V^T tiles [ch128][key32]
  __shared__ unsigned short lp[4][16 * 32];     // per-wave P tile [qrow16][key32]

  const int t    = threadIdx.x;
  const int lane = t & 31;
  const int w    = t >> 5;
  const int mloc = lane & 15;
  const int hi   = lane >> 4;

  const int bh = blockIdx.y;
  const int b  = bh >> 2;
  const int h  = bh & 3;
  const int qbase = blockIdx.x * 64 + w * 16;

  const unsigned short* Qb  = Qm  + (size_t)bh * Ndim * DK;
  const unsigned short* Kb  = Km  + (size_t)bh * Ndim * DK;
  const unsigned short* Vtb = Vtm + (size_t)bh * DK * Ndim;
  const int* adjb = adjs + (size_t)b * Ndim * Ndim;

  // resident Q fragments: 16 rows x 128 ch = 4 A-frags
  BF16Frag qf[4];
  for (int c = 0; c < 4; ++c)
    for (int p = 0; p < 8; ++p) {
      const int kk = c * 32 + a_pair_k(p, hi);
      qf[c].u[p] = *(const unsigned*)(Qb + (size_t)(qbase + mloc) * DK + kk);
    }

  // all-ones bf16 B fragment: rowsum(P) via the matrix pipe
  BF16Frag ones;
  for (int p = 0; p < 8; ++p) ones.u[p] = 0x3F803F80u;

  F32Acc o[8], lacc;
  for (int d = 0; d < 8; ++d)
    for (int r = 0; r < 8; ++r) o[d].f[r] = 0.0f;
  for (int r = 0; r < 8; ++r) lacc.f[r] = 0.0f;
  float mrow[8];
  for (int r = 0; r < 8; ++r) mrow[r] = -INFINITY;

  // (1/sqrt(512)) * log2(e): softmax done in exp2
  const float sc = 0.044194173824159216f * 1.4426950408889634f;

  stage_tiles(0, lk[0], lvt[0], Kb, Vtb, t);   // prologue: tile 0 in flight

  for (int j = 0; j < 32; ++j) {
    const int kb  = j * 32;
    const int cur = j & 1;
    const unsigned short* lkc = lk[cur];
    const unsigned short* lvc = lvt[cur];

    if (j + 1 < 32) {
      // other buffer was fully consumed at iter j-1's trailing barrier
      stage_tiles(kb + 32, lk[cur ^ 1], lvt[cur ^ 1], Kb, Vtb, t);
#ifdef USE_ASYNC_LDS
      __builtin_amdgcn_s_wait_asynccnt(8);  // in-order: tile j resident, j+1 in flight
#endif
    } else {
#ifdef USE_ASYNC_LDS
      __builtin_amdgcn_s_wait_asynccnt(0);
#endif
    }
    __syncthreads();

    // S(16x32) = Q K^T as two 16x16 WMMA chains over 128 channels
    F32Acc s0, s1;
    for (int r = 0; r < 8; ++r) { s0.f[r] = 0.0f; s1.f[r] = 0.0f; }
    for (int c = 0; c < 4; ++c) {
      BF16Frag b0, b1;
      for (int p = 0; p < 8; ++p) {
        const int kk = c * 32 + hi * 16 + 2 * p;
        b0.u[p] = ld_u32(&lkc[mloc * 128 + kk]);
        b1.u[p] = ld_u32(&lkc[(mloc + 16) * 128 + kk]);
      }
      s0.v = __builtin_amdgcn_wmma_f32_16x16x32_bf16(false, qf[c].v, false, b0.v,
                                                     (short)0, s0.v, false, false);
      s1.v = __builtin_amdgcn_wmma_f32_16x16x32_bf16(false, qf[c].v, false, b1.v,
                                                     (short)0, s1.v, false, false);
    }

    // adjacency mask + scale + online softmax (rows live in 16-lane halves)
    for (int r = 0; r < 8; ++r) {
      const int ml  = hi ? 8 + r : r;
      const int row = qbase + ml;
      const int* arow = adjb + (size_t)row * Ndim + kb;
      float x0 = (arow[mloc] != 0)      ? s0.f[r] * sc : -INFINITY;
      float x1 = (arow[mloc + 16] != 0) ? s1.f[r] * sc : -INFINITY;
      float vmax = fmaxf(x0, x1);
      for (int off = 1; off < 16; off <<= 1)
        vmax = fmaxf(vmax, __shfl_xor(vmax, off, 32));
      const float nm = fmaxf(mrow[r], vmax);
      float p0, p1, alpha;
      if (nm == -INFINITY) { p0 = 0.0f; p1 = 0.0f; alpha = 1.0f; }
      else {
        p0 = exp2f(x0 - nm);
        p1 = exp2f(x1 - nm);
        alpha = (mrow[r] == -INFINITY) ? 0.0f : exp2f(mrow[r] - nm);
      }
      mrow[r] = nm;
      lacc.f[r] *= alpha;
      for (int d = 0; d < 8; ++d) o[d].f[r] *= alpha;
      // bounce P through LDS: C-layout -> A-layout
      lp[w][ml * 32 + mloc]      = f2bf(p0);
      lp[w][ml * 32 + mloc + 16] = f2bf(p1);
    }

    // O += P(16x32) @ V(32x128);  l += P @ ones (row sums on the matrix pipe)
    BF16Frag pf;
    for (int p = 0; p < 8; ++p)
      pf.u[p] = ld_u32(&lp[w][mloc * 32 + a_pair_k(p, hi)]);
    lacc.v = __builtin_amdgcn_wmma_f32_16x16x32_bf16(false, pf.v, false, ones.v,
                                                     (short)0, lacc.v, false, false);
    for (int d = 0; d < 8; ++d) {
      BF16Frag vf;
      for (int p = 0; p < 8; ++p)
        vf.u[p] = ld_u32(&lvc[(d * 16 + mloc) * 32 + hi * 16 + 2 * p]);
      o[d].v = __builtin_amdgcn_wmma_f32_16x16x32_bf16(false, pf.v, false, vf.v,
                                                       (short)0, o[d].v, false, false);
    }
    __syncthreads();
  }

  // epilogue: normalize, ReLU, store fp32 (concat heads along feature dim)
  for (int r = 0; r < 8; ++r) {
    const int row = qbase + (hi ? 8 + r : r);
    const float l = lacc.f[r];
    const float inv = (l > 0.0f) ? (1.0f / l) : 0.0f;
    float* orow = out + ((size_t)b * Ndim + row) * Hdim + h * DK;
    for (int d = 0; d < 8; ++d)
      orow[d * 16 + mloc] = fmaxf(o[d].f[r] * inv, 0.0f);
  }
}

// ---------------------------------------------------------------------------
extern "C" void kernel_launch(void* const* d_in, const int* in_sizes, int n_in,
                              void* d_out, int out_size, void* d_ws, size_t ws_size,
                              hipStream_t stream) {
  const int*   adjs      = (const int*)d_in[0];
  const float* nodes     = (const float*)d_in[2];
  const float* relations = (const float*)d_in[4];
  const float* Wq        = (const float*)d_in[5];
  const float* Wk        = (const float*)d_in[6];
  const float* Wv        = (const float*)d_in[7];
  float* out = (float*)d_out;

  // workspace: Q,K row-major bf16; V transposed bf16 (8 MB each, 24 MB total)
  unsigned short* Q  = (unsigned short*)d_ws;
  unsigned short* K  = Q + (size_t)Bdim * 4 * Ndim * DK;
  unsigned short* Vt = K + (size_t)Bdim * 4 * Ndim * DK;

  dim3 gp(128, 8);  // row tiles, col tiles
  hipLaunchKernelGGL(qkv_proj_kernel<0>, gp, dim3(128), 0, stream, nodes, Wq, Q);
  hipLaunchKernelGGL(qkv_proj_kernel<1>, gp, dim3(128), 0, stream, nodes, Wk, K);
  hipLaunchKernelGGL(qkv_proj_kernel<2>, gp, dim3(128), 0, stream, nodes, Wv, Vt);

  dim3 ga(16, 32);  // query tiles, B*HEADS
  hipLaunchKernelGGL(gat_attn_kernel, ga, dim3(128), 0, stream, Q, K, Vt, adjs, out);

  // relations passthrough (second tuple element)
  (void)hipMemcpyAsync(out + (size_t)Bdim * Ndim * Hdim, relations,
                       (size_t)Bdim * 64 * Hdim * sizeof(float),
                       hipMemcpyDeviceToDevice, stream);
}
